// ExecutionEncoder_57552561766403
// MI455X (gfx1250) — compile-verified, matrix operand order checked by reading, not run on hardware
//
#include <hip/hip_runtime.h>
#include <hip/hip_bf16.h>
#include <math.h>

// ---------------- model constants (match reference) ----------------
#define HID     512
#define HEADS   8
#define DH      64
#define EDIM    64
#define NLAYERS 4
#define LAT     1024
#define FFND    (4 * HID)

// ---------------- WMMA fragment types (CDNA5 wave32) ----------------
typedef __bf16          v16bf  __attribute__((ext_vector_type(16)));
typedef float           v8f    __attribute__((ext_vector_type(8)));
typedef unsigned short  u16x16 __attribute__((ext_vector_type(16)));
typedef unsigned short  u16x8  __attribute__((ext_vector_type(8)));

static __device__ __forceinline__ unsigned short f2bf(float f) {
  unsigned u = __builtin_bit_cast(unsigned, f);
  unsigned r = u + 0x7fffu + ((u >> 16) & 1u);   // round-to-nearest-even
  return (unsigned short)(r >> 16);
}
static __device__ __forceinline__ float gelu_exact(float x) {
  return 0.5f * x * (1.0f + erff(x * 0.70710678118654752f));
}
// order-preserving float<->uint for atomic max on signed floats
static __device__ __forceinline__ unsigned enc_f(float f) {
  unsigned u = __builtin_bit_cast(unsigned, f);
  return (u & 0x80000000u) ? ~u : (u | 0x80000000u);
}
static __device__ __forceinline__ float dec_f(unsigned u) {
  unsigned v = (u & 0x80000000u) ? (u & 0x7fffffffu) : ~u;
  return __builtin_bit_cast(float, v);
}

static __device__ __forceinline__ v8f wmma_bf16(v16bf a, v16bf b, v8f c) {
  return __builtin_amdgcn_wmma_f32_16x16x32_bf16(false, a, false, b, (short)0, c,
                                                 false, false);
}

// 16-bit A/B fragment: 16 bf16/lane = two contiguous 8-element runs at b0, b1
// (ISA 7.12.2 16-bit layout: VGPR v<4 -> K = b0+0..7, v>=4 -> K = b1+0..7)
static __device__ __forceinline__ v16bf load_frag(const unsigned short* __restrict__ p,
                                                  int b0, int b1) {
  u16x8 y0 = *(const u16x8*)(p + b0);
  u16x8 y1 = *(const u16x8*)(p + b1);
  u16x16 u = __builtin_shufflevector(y0, y1, 0, 1, 2, 3, 4, 5, 6, 7,
                                             8, 9, 10, 11, 12, 13, 14, 15);
  return __builtin_bit_cast(v16bf, u);
}

// ---------------- converts ----------------
// f32 (K x N) -> bf16 transposed (N x K)
__global__ void k_cvt_wT(const float* __restrict__ w, unsigned short* __restrict__ wT,
                         int K, int N) {
  size_t i = (size_t)blockIdx.x * blockDim.x + threadIdx.x;
  size_t total = (size_t)K * N;
  if (i >= total) return;
  int n = (int)(i / K), k = (int)(i % K);
  wT[i] = f2bf(w[(size_t)k * N + n]);
}
// f32 -> bf16, same layout
__global__ void k_cvt_bf16(const float* __restrict__ s, unsigned short* __restrict__ d,
                           size_t n) {
  size_t i = (size_t)blockIdx.x * blockDim.x + threadIdx.x;
  if (i < n) d[i] = f2bf(s[i]);
}

// ---------------- generic WMMA GEMM: C = epi(A @ W + bias) ----------------
// A operand is bf16 row-major (M x K).
// MODE 0: A given directly.
// MODE 1: A[i][k] = (k < Kh) ? T0[idx0[i*stride]*Kh + k] : T1[idx1[i*stride]*Kh + k-Kh]
// WT: bf16, N x K (transposed weight). EPI: 0 none, 1 exact GELU.
// OUTBF: 0 -> f32 C, 1 -> bf16 C.
// Wave tile 32x32 (2x2 WMMA register block); 8 waves/block -> 128 x 64 block tile.
template <int MODE, int EPI, int OUTBF>
__global__ void k_wmma_gemm(
    const unsigned short* __restrict__ A,
    const unsigned short* __restrict__ T0, const unsigned short* __restrict__ T1,
    const int* __restrict__ idx0, const int* __restrict__ idx1,
    int idxStride, int Kh,
    const unsigned short* __restrict__ WT,
    const float* __restrict__ bias,
    void* __restrict__ Cv,
    int M, int N, int K)
{
  const int lane = threadIdx.x & 31;
  const int wave = threadIdx.x >> 5;
  const int m16  = lane & 15;      // row-in-tile (A) / col-in-tile (B,D)
  const int half = lane >> 4;      // K-half selector for A/B fragments
  const int tm   = blockIdx.y * 4 + (wave >> 1);   // 32-row wave tile index
  const int tn   = blockIdx.x * 2 + (wave & 1);    // 32-col wave tile index
  if (tm * 32 >= M || tn * 32 >= N) return;

  int rowA0 = tm * 32 + m16;      if (rowA0 >= M) rowA0 = M - 1;
  int rowA1 = tm * 32 + 16 + m16; if (rowA1 >= M) rowA1 = M - 1;

  const unsigned short *ar0 = nullptr, *ar1 = nullptr;
  const unsigned short *g00 = nullptr, *g01 = nullptr, *g10 = nullptr, *g11 = nullptr;
  if (MODE == 0) {
    ar0 = A + (size_t)rowA0 * K;
    ar1 = A + (size_t)rowA1 * K;
  } else {
    g00 = T0 + (size_t)idx0[(size_t)rowA0 * idxStride] * Kh;
    g01 = T1 + (size_t)idx1[(size_t)rowA0 * idxStride] * Kh - Kh;
    g10 = T0 + (size_t)idx0[(size_t)rowA1 * idxStride] * Kh;
    g11 = T1 + (size_t)idx1[(size_t)rowA1 * idxStride] * Kh - Kh;
  }
  int colB0 = tn * 32 + m16;      if (colB0 >= N) colB0 = N - 1;
  int colB1 = tn * 32 + 16 + m16; if (colB1 >= N) colB1 = N - 1;
  const unsigned short* br0 = WT + (size_t)colB0 * K;
  const unsigned short* br1 = WT + (size_t)colB1 * K;

  v8f acc00 = {}, acc01 = {}, acc10 = {}, acc11 = {};

  for (int k0 = 0; k0 < K; k0 += 32) {
    const int b0 = k0 + 8 * half;
    const int b1 = k0 + 16 + 8 * half;
    const unsigned short *pa0, *pa1;
    if (MODE == 0) { pa0 = ar0; pa1 = ar1; }
    else {         // Kh is a multiple of 32 -> whole chunk on one side of concat
      pa0 = (k0 < Kh) ? g00 : g01;
      pa1 = (k0 < Kh) ? g10 : g11;
    }
    if (k0 + 128 < K) {                            // stream-ahead hints
      __builtin_prefetch(br0 + k0 + 128, 0, 1);
      __builtin_prefetch(br1 + k0 + 128, 0, 1);
      __builtin_prefetch(pa0 + b0 + 128, 0, 1);
      __builtin_prefetch(pa1 + b0 + 128, 0, 1);
    }
    v16bf a0  = load_frag(pa0, b0, b1);
    v16bf a1  = load_frag(pa1, b0, b1);
    v16bf bb0 = load_frag(br0, b0, b1);
    v16bf bb1 = load_frag(br1, b0, b1);
    acc00 = wmma_bf16(a0, bb0, acc00);
    acc01 = wmma_bf16(a0, bb1, acc01);
    acc10 = wmma_bf16(a1, bb0, acc10);
    acc11 = wmma_bf16(a1, bb1, acc11);
  }

  float* Cf = (float*)Cv;
  unsigned short* Cb = (unsigned short*)Cv;
  const int c0 = tn * 32 + m16, c1 = tn * 32 + 16 + m16;
  const float bv0 = (bias && c0 < N) ? bias[c0] : 0.0f;
  const float bv1 = (bias && c1 < N) ? bias[c1] : 0.0f;
  const int r0 = tm * 32, r1 = tm * 32 + 16;
#pragma unroll
  for (int r = 0; r < 8; ++r) {
    const int rowa = r0 + r + 8 * half;     // D layout: VGPR r -> M=r (+8 for hi lanes)
    const int rowb = r1 + r + 8 * half;
    if (rowa < M) {
      if (c0 < N) {
        float o = acc00[r] + bv0; if (EPI == 1) o = gelu_exact(o);
        if (OUTBF) Cb[(size_t)rowa * N + c0] = f2bf(o); else Cf[(size_t)rowa * N + c0] = o;
      }
      if (c1 < N) {
        float o = acc01[r] + bv1; if (EPI == 1) o = gelu_exact(o);
        if (OUTBF) Cb[(size_t)rowa * N + c1] = f2bf(o); else Cf[(size_t)rowa * N + c1] = o;
      }
    }
    if (rowb < M) {
      if (c0 < N) {
        float o = acc10[r] + bv0; if (EPI == 1) o = gelu_exact(o);
        if (OUTBF) Cb[(size_t)rowb * N + c0] = f2bf(o); else Cf[(size_t)rowb * N + c0] = o;
      }
      if (c1 < N) {
        float o = acc11[r] + bv1; if (EPI == 1) o = gelu_exact(o);
        if (OUTBF) Cb[(size_t)rowb * N + c1] = f2bf(o); else Cf[(size_t)rowb * N + c1] = o;
      }
    }
  }
}

// ---------------- LayerNorm over rows of width W; OUTBF selects f32/bf16 out ----------------
template <int OUTBF>
__global__ void k_ln(const float* __restrict__ x, const float* __restrict__ g,
                     const float* __restrict__ b, void* __restrict__ yv, int W) {
  const int row = blockIdx.x;
  const float* xr = x + (size_t)row * W;
  float s = 0.f, s2 = 0.f;
  for (int c = threadIdx.x; c < W; c += blockDim.x) { float v = xr[c]; s += v; s2 += v * v; }
  __shared__ float ls[16], ls2[16];
  for (int off = 16; off; off >>= 1) { s += __shfl_down(s, off, 32); s2 += __shfl_down(s2, off, 32); }
  const int lane = threadIdx.x & 31, wv = threadIdx.x >> 5, nw = blockDim.x >> 5;
  if (lane == 0) { ls[wv] = s; ls2[wv] = s2; }
  __syncthreads();
  if (wv == 0) {
    s  = (lane < nw) ? ls[lane]  : 0.f;
    s2 = (lane < nw) ? ls2[lane] : 0.f;
    for (int off = 16; off; off >>= 1) { s += __shfl_down(s, off, 32); s2 += __shfl_down(s2, off, 32); }
    if (lane == 0) { ls[0] = s; ls2[0] = s2; }
  }
  __syncthreads();
  const float mu  = ls[0] / W;
  const float var = ls2[0] / W - mu * mu;
  const float inv = rsqrtf(var + 1e-5f);
  float* yf = (float*)yv;
  unsigned short* yb = (unsigned short*)yv;
  for (int c = threadIdx.x; c < W; c += blockDim.x) {
    float o = (xr[c] - mu) * inv * g[c] + b[c];
    if (OUTBF) yb[(size_t)row * W + c] = f2bf(o);
    else       yf[(size_t)row * W + c] = o;
  }
}

// ---------------- embedding / elementwise ----------------
__global__ void k_embed(const int* __restrict__ tool_ids,
                        const float* __restrict__ emb_tool,
                        const float* __restrict__ emb_pos,
                        float* __restrict__ x, int Nn) {
  size_t i = (size_t)blockIdx.x * blockDim.x + threadIdx.x;
  if (i >= (size_t)Nn * HID) return;
  int node = (int)(i / HID), c = (int)(i % HID);
  x[i] = emb_tool[(size_t)tool_ids[node] * HID + c] + emb_pos[i];
}
__global__ void k_add2(float* __restrict__ x, const float* __restrict__ a,
                       const float* __restrict__ b, size_t n) {
  size_t i = (size_t)blockIdx.x * blockDim.x + threadIdx.x;
  if (i < n) x[i] += a[i] + b[i];
}
__global__ void k_add1(float* __restrict__ x, const float* __restrict__ a, size_t n) {
  size_t i = (size_t)blockIdx.x * blockDim.x + threadIdx.x;
  if (i < n) x[i] += a[i];
}
__global__ void k_fill_f32(float* p, float v, size_t n) {
  size_t i = (size_t)blockIdx.x * blockDim.x + threadIdx.x;
  if (i < n) p[i] = v;
}
__global__ void k_fill_u32(unsigned* p, unsigned v, size_t n) {
  size_t i = (size_t)blockIdx.x * blockDim.x + threadIdx.x;
  if (i < n) p[i] = v;
}

// ---------------- per-edge attention scores ----------------
// s[e,h] = (q[tgt]·k[src])_h / sqrt(DH) + t[e]·ea_w[:,h] + ea_b[h]
__global__ void k_scores(const float* __restrict__ q, const float* __restrict__ k,
                         const float* __restrict__ t,
                         const float* __restrict__ ea_w, const float* __restrict__ ea_b,
                         const int* __restrict__ edges, float* __restrict__ s, int E) {
  int i = blockIdx.x * blockDim.x + threadIdx.x;
  if (i >= E * HEADS) return;
  int e = i >> 3, h = i & 7;
  int src = edges[2 * e], tgt = edges[2 * e + 1];
  const float* qp = q + (size_t)tgt * HID + h * DH;
  const float* kp = k + (size_t)src * HID + h * DH;
  float d = 0.f;
#pragma unroll 8
  for (int j = 0; j < DH; ++j) d += qp[j] * kp[j];
  d *= 0.125f;                                    // DH^-0.5
  float a = ea_b[h];
  const float* tp = t + (size_t)e * EDIM;
#pragma unroll 8
  for (int j = 0; j < EDIM; ++j) a += tp[j] * ea_w[j * HEADS + h];
  s[i] = d + a;
}

// ---------------- segment softmax over tgt + weighted scatter of v[src] ----------------
__global__ void k_segmax(const float* __restrict__ s, const int* __restrict__ edges,
                         unsigned* __restrict__ mbuf, int E) {
  int i = blockIdx.x * blockDim.x + threadIdx.x;
  if (i >= E * HEADS) return;
  int e = i >> 3, h = i & 7;
  atomicMax(&mbuf[(size_t)edges[2 * e + 1] * HEADS + h], enc_f(s[i]));
}
__global__ void k_expsum(float* __restrict__ s, const int* __restrict__ edges,
                         const unsigned* __restrict__ mbuf, float* __restrict__ den, int E) {
  int i = blockIdx.x * blockDim.x + threadIdx.x;
  if (i >= E * HEADS) return;
  int e = i >> 3, h = i & 7;
  int tgt = edges[2 * e + 1];
  float ex = expf(s[i] - dec_f(mbuf[(size_t)tgt * HEADS + h]));
  s[i] = ex;
  atomicAdd(&den[(size_t)tgt * HEADS + h], ex);
}
__global__ void k_scatter(const float* __restrict__ s, const float* __restrict__ den,
                          const float* __restrict__ v, const int* __restrict__ edges,
                          float* __restrict__ attn, int E) {
  size_t i = (size_t)blockIdx.x * blockDim.x + threadIdx.x;
  if (i >= (size_t)E * HID) return;
  int e = (int)(i / HID), c = (int)(i % HID);
  int h = c >> 6;
  int src = edges[2 * e], tgt = edges[2 * e + 1];
  float w = s[(size_t)e * HEADS + h] / den[(size_t)tgt * HEADS + h];
  atomicAdd(&attn[(size_t)tgt * HID + c], w * v[(size_t)src * HID + c]);
}

// ---------------- pooling + head ----------------
__global__ void k_logits(const float* __restrict__ x, const float* __restrict__ pw,
                         const float* __restrict__ pb, float* __restrict__ logits, int Nn) {
  int n = blockIdx.x * blockDim.x + threadIdx.x;
  if (n >= Nn) return;
  const float* xr = x + (size_t)n * HID;
  float a = pb[0];
#pragma unroll 8
  for (int c = 0; c < HID; ++c) a += xr[c] * pw[c];
  logits[n] = a;
}
__global__ void k_redmax(const float* __restrict__ logits, unsigned* __restrict__ r, int Nn) {
  int n = blockIdx.x * blockDim.x + threadIdx.x;
  if (n < Nn) atomicMax(r, enc_f(logits[n]));
}
__global__ void k_redsum(const float* __restrict__ logits, const unsigned* __restrict__ rmax,
                         float* __restrict__ rsum, int Nn) {
  int n = blockIdx.x * blockDim.x + threadIdx.x;
  if (n < Nn) atomicAdd(rsum, expf(logits[n] - dec_f(rmax[0])));
}
__global__ void k_pool(const float* __restrict__ x, const float* __restrict__ logits,
                       const unsigned* __restrict__ rmax, const float* __restrict__ rsum,
                       float* __restrict__ pooled, int Nn) {
  __shared__ float acc[HID];
  for (int c = threadIdx.x; c < HID; c += blockDim.x) acc[c] = 0.f;
  __syncthreads();
  const float mx = dec_f(rmax[0]);
  const float inv = 1.f / rsum[0];
  int n0 = blockIdx.x * 64, n1 = n0 + 64; if (n1 > Nn) n1 = Nn;
  for (int n = n0; n < n1; ++n) {
    float p = expf(logits[n] - mx) * inv;
    const float* xr = x + (size_t)n * HID;
    for (int c = threadIdx.x; c < HID; c += blockDim.x) acc[c] += p * xr[c];
  }
  __syncthreads();
  for (int c = threadIdx.x; c < HID; c += blockDim.x) atomicAdd(&pooled[c], acc[c]);
}
__global__ void k_mlp1(const float* __restrict__ pooled, const float* __restrict__ w,
                       const float* __restrict__ b, float* __restrict__ h1) {
  int j = blockIdx.x * blockDim.x + threadIdx.x;
  if (j >= 2 * LAT) return;
  float a = b[j];
  for (int i = 0; i < HID; ++i) a += pooled[i] * w[(size_t)i * (2 * LAT) + j];
  h1[j] = gelu_exact(a);
}
__global__ void k_mlp2(const float* __restrict__ h1, const float* __restrict__ w,
                       const float* __restrict__ b, float* __restrict__ z) {
  int j = blockIdx.x * blockDim.x + threadIdx.x;
  if (j >= LAT) return;
  float a = b[j];
  for (int i = 0; i < 2 * LAT; ++i) a += h1[i] * w[(size_t)i * LAT + j];
  z[j] = a;
}

// =====================================================================
extern "C" void kernel_launch(void* const* d_in, const int* in_sizes, int n_in,
                              void* d_out, int out_size, void* d_ws, size_t ws_size,
                              hipStream_t stream) {
  (void)n_in; (void)out_size; (void)ws_size;
  // ---- inputs, flattened in setup_inputs() / params dict order ----
  const int*   tool_ids  = (const int*)  d_in[0];
  const int*   tiers     = (const int*)  d_in[1];
  const int*   scopes    = (const int*)  d_in[2];
  const float* arg_emb   = (const float*)d_in[3];
  const int*   edges     = (const int*)  d_in[4];
  const float* emb_tool  = (const float*)d_in[5];
  const float* emb_pos   = (const float*)d_in[6];
  const float* emb_tier  = (const float*)d_in[7];
  const float* emb_scope = (const float*)d_in[8];
  const float* fus_w     = (const float*)d_in[9];
  const float* fus_b     = (const float*)d_in[10];
  const float* arg_w     = (const float*)d_in[11];
  const float* arg_b     = (const float*)d_in[12];
  const float* in_g      = (const float*)d_in[13];
  const float* in_b      = (const float*)d_in[14];
  const float* q_w  = (const float*)d_in[15]; const float* q_b  = (const float*)d_in[16];
  const float* k_w  = (const float*)d_in[17]; const float* k_b  = (const float*)d_in[18];
  const float* v_w  = (const float*)d_in[19]; const float* v_b  = (const float*)d_in[20];
  const float* o_w  = (const float*)d_in[21]; const float* o_b  = (const float*)d_in[22];
  const float* ep_w = (const float*)d_in[23]; const float* ep_b = (const float*)d_in[24];
  const float* ea_w = (const float*)d_in[25]; const float* ea_b = (const float*)d_in[26];
  const float* n1_g = (const float*)d_in[27]; const float* n1_b = (const float*)d_in[28];
  const float* n2_g = (const float*)d_in[29]; const float* n2_b = (const float*)d_in[30];
  const float* f1_w = (const float*)d_in[31]; const float* f1_b = (const float*)d_in[32];
  const float* f2_w = (const float*)d_in[33]; const float* f2_b = (const float*)d_in[34];
  const float* pool_w = (const float*)d_in[35]; const float* pool_b = (const float*)d_in[36];
  const float* p1_w   = (const float*)d_in[37]; const float* p1_b   = (const float*)d_in[38];
  const float* p2_w   = (const float*)d_in[39]; const float* p2_b   = (const float*)d_in[40];
  const float* pn_g   = (const float*)d_in[41]; const float* pn_b   = (const float*)d_in[42];

  const int Nn = in_sizes[0];        // 10000
  const int E  = in_sizes[4] / 2;    // 170000

  // ---- workspace carve-up ----
  unsigned char* wp = (unsigned char*)d_ws;
  auto alloc = [&](size_t bytes) -> unsigned char* {
    unsigned char* p = wp; wp += (bytes + 255) & ~(size_t)255; return p;
  };
  float* x     = (float*)alloc((size_t)Nn * HID * 4);
  float* qb    = (float*)alloc((size_t)Nn * HID * 4);
  float* kb    = (float*)alloc((size_t)Nn * HID * 4);
  float* vb    = (float*)alloc((size_t)Nn * HID * 4);
  float* attn  = (float*)alloc((size_t)Nn * HID * 4);
  float* ao    = (float*)alloc((size_t)Nn * HID * 4);
  float* prov  = (float*)alloc((size_t)Nn * HID * 4);
  float* argp  = (float*)alloc((size_t)Nn * HID * 4);
  float* tbuf  = (float*)alloc((size_t)E * EDIM * 4);
  float* sbuf  = (float*)alloc((size_t)E * HEADS * 4);
  unsigned* mbuf = (unsigned*)alloc((size_t)Nn * HEADS * 4);
  float* den   = (float*)alloc((size_t)Nn * HEADS * 4);
  float* logits = (float*)alloc((size_t)Nn * 4);
  unsigned* rmax = (unsigned*)alloc(4);
  float* rsum  = (float*)alloc(4);
  float* pooled = (float*)alloc(HID * 4);
  float* h1    = (float*)alloc(2 * LAT * 4);
  float* zbuf  = (float*)alloc(LAT * 4);
  // bf16 activation buffers (GEMM A operands)
  unsigned short* xnb   = (unsigned short*)alloc((size_t)Nn * HID * 2);
  unsigned short* ffnb  = (unsigned short*)alloc((size_t)Nn * FFND * 2);
  unsigned short* attnb = (unsigned short*)alloc((size_t)Nn * HID * 2);
  unsigned short* argb  = (unsigned short*)alloc((size_t)Nn * HID * 2);
  unsigned short* tierb  = (unsigned short*)alloc((size_t)3 * HID * 2);
  unsigned short* scopeb = (unsigned short*)alloc((size_t)10 * HID * 2);
  // bf16 transposed weights
  unsigned short* fusT = (unsigned short*)alloc((size_t)2 * HID * HID * 2);
  unsigned short* argT = (unsigned short*)alloc((size_t)HID * HID * 2);
  unsigned short *qT[NLAYERS], *kT[NLAYERS], *vT[NLAYERS], *oT[NLAYERS],
                 *epT[NLAYERS], *f1T[NLAYERS], *f2T[NLAYERS];
  for (int L = 0; L < NLAYERS; ++L) {
    qT[L]  = (unsigned short*)alloc((size_t)HID * HID * 2);
    kT[L]  = (unsigned short*)alloc((size_t)HID * HID * 2);
    vT[L]  = (unsigned short*)alloc((size_t)HID * HID * 2);
    oT[L]  = (unsigned short*)alloc((size_t)HID * HID * 2);
    epT[L] = (unsigned short*)alloc((size_t)2 * HID * EDIM * 2);
    f1T[L] = (unsigned short*)alloc((size_t)HID * FFND * 2);
    f2T[L] = (unsigned short*)alloc((size_t)FFND * HID * 2);
  }

  auto grid1 = [](size_t n) { return dim3((unsigned)((n + 255) / 256)); };
  auto cvt = [&](const float* w, unsigned short* wT, int K, int N) {
    k_cvt_wT<<<grid1((size_t)K * N), 256, 0, stream>>>(w, wT, K, N);
  };
  auto cvtb = [&](const float* s, unsigned short* d, size_t n) {
    k_cvt_bf16<<<grid1(n), 256, 0, stream>>>(s, d, n);
  };
  auto gemm = [&](const unsigned short* A, const unsigned short* WT, const float* bias,
                  void* C, int M, int N, int K, int epi, int outbf) {
    dim3 g((N + 63) / 64, (M + 127) / 128);
    if (epi == 0)
      k_wmma_gemm<0, 0, 0><<<g, 256, 0, stream>>>(A, nullptr, nullptr, nullptr, nullptr,
                                                  1, 0, WT, bias, C, M, N, K);
    else if (outbf)
      k_wmma_gemm<0, 1, 1><<<g, 256, 0, stream>>>(A, nullptr, nullptr, nullptr, nullptr,
                                                  1, 0, WT, bias, C, M, N, K);
    else
      k_wmma_gemm<0, 1, 0><<<g, 256, 0, stream>>>(A, nullptr, nullptr, nullptr, nullptr,
                                                  1, 0, WT, bias, C, M, N, K);
  };
  auto gemm_gather = [&](const unsigned short* T0, const unsigned short* T1,
                         const int* i0, const int* i1, int istride, int Kh,
                         const unsigned short* WT, const float* bias,
                         float* C, int M, int N, int K) {
    dim3 g((N + 63) / 64, (M + 127) / 128);
    k_wmma_gemm<1, 0, 0><<<g, 256, 0, stream>>>(nullptr, T0, T1, i0, i1, istride, Kh,
                                                WT, bias, C, M, N, K);
  };

  // ---- one-time conversions: weights (transposed) + constant tables ----
  cvt(fus_w, fusT, 2 * HID, HID);
  cvt(arg_w, argT, HID, HID);
  cvtb(emb_tier, tierb, (size_t)3 * HID);
  cvtb(emb_scope, scopeb, (size_t)10 * HID);
  cvtb(arg_emb, argb, (size_t)Nn * HID);
  for (int L = 0; L < NLAYERS; ++L) {
    cvt(q_w  + (size_t)L * HID * HID,  qT[L],  HID, HID);
    cvt(k_w  + (size_t)L * HID * HID,  kT[L],  HID, HID);
    cvt(v_w  + (size_t)L * HID * HID,  vT[L],  HID, HID);
    cvt(o_w  + (size_t)L * HID * HID,  oT[L],  HID, HID);
    cvt(ep_w + (size_t)L * 2 * HID * EDIM, epT[L], 2 * HID, EDIM);
    cvt(f1_w + (size_t)L * HID * FFND, f1T[L], HID, FFND);
    cvt(f2_w + (size_t)L * FFND * HID, f2T[L], FFND, HID);
  }

  // ---- input embedding ----
  k_embed<<<grid1((size_t)Nn * HID), 256, 0, stream>>>(tool_ids, emb_tool, emb_pos, x, Nn);
  gemm_gather(tierb, scopeb, tiers, scopes, 1, HID, fusT, fus_b, prov, Nn, HID, 2 * HID);
  gemm(argb, argT, arg_b, argp, Nn, HID, HID, 1, 0);                 // GELU epilogue
  k_add2<<<grid1((size_t)Nn * HID), 256, 0, stream>>>(x, prov, argp, (size_t)Nn * HID);
  k_ln<0><<<Nn, 256, 0, stream>>>(x, in_g, in_b, x, HID);

  // ---- GAT layers ----
  for (int L = 0; L < NLAYERS; ++L) {
    // ln1 -> bf16 activations (feeds q/k/v + edge-feature GEMMs)
    k_ln<1><<<Nn, 256, 0, stream>>>(x, n1_g + (size_t)L * HID, n1_b + (size_t)L * HID,
                                    xnb, HID);
    gemm(xnb, qT[L], q_b + (size_t)L * HID, qb, Nn, HID, HID, 0, 0);
    gemm(xnb, kT[L], k_b + (size_t)L * HID, kb, Nn, HID, HID, 0, 0);
    gemm(xnb, vT[L], v_b + (size_t)L * HID, vb, Nn, HID, HID, 0, 0);
    // edge features: concat(xn[src], xn[tgt]) @ ep_w, never materialized
    gemm_gather(xnb, xnb, edges + 0, edges + 1, 2, HID,
                epT[L], ep_b + (size_t)L * EDIM, tbuf, E, EDIM, 2 * HID);
    k_scores<<<grid1((size_t)E * HEADS), 256, 0, stream>>>(
        qb, kb, tbuf, ea_w + (size_t)L * EDIM * HEADS, ea_b + (size_t)L * HEADS,
        edges, sbuf, E);
    k_fill_u32<<<grid1((size_t)Nn * HEADS), 256, 0, stream>>>(mbuf, 0u, (size_t)Nn * HEADS);
    k_segmax<<<grid1((size_t)E * HEADS), 256, 0, stream>>>(sbuf, edges, mbuf, E);
    k_fill_f32<<<grid1((size_t)Nn * HEADS), 256, 0, stream>>>(den, 0.f, (size_t)Nn * HEADS);
    k_expsum<<<grid1((size_t)E * HEADS), 256, 0, stream>>>(sbuf, edges, mbuf, den, E);
    k_fill_f32<<<grid1((size_t)Nn * HID), 256, 0, stream>>>(attn, 0.f, (size_t)Nn * HID);
    k_scatter<<<grid1((size_t)E * HID), 256, 0, stream>>>(sbuf, den, vb, edges, attn, E);
    cvtb(attn, attnb, (size_t)Nn * HID);
    gemm(attnb, oT[L], o_b + (size_t)L * HID, ao, Nn, HID, HID, 0, 0);
    k_add1<<<grid1((size_t)Nn * HID), 256, 0, stream>>>(x, ao, (size_t)Nn * HID);
    // FFN
    k_ln<1><<<Nn, 256, 0, stream>>>(x, n2_g + (size_t)L * HID, n2_b + (size_t)L * HID,
                                    xnb, HID);
    gemm(xnb, f1T[L], f1_b + (size_t)L * FFND, ffnb, Nn, FFND, HID, 1, 1);  // GELU->bf16
    gemm(ffnb, f2T[L], f2_b + (size_t)L * HID, ao, Nn, HID, FFND, 0, 0);
    k_add1<<<grid1((size_t)Nn * HID), 256, 0, stream>>>(x, ao, (size_t)Nn * HID);
  }

  // ---- attention pooling + latent head ----
  k_logits<<<grid1((size_t)Nn), 256, 0, stream>>>(x, pool_w, pool_b, logits, Nn);
  k_fill_u32<<<1, 1, 0, stream>>>(rmax, 0u, 1);
  k_redmax<<<grid1((size_t)Nn), 256, 0, stream>>>(logits, rmax, Nn);
  k_fill_f32<<<1, 1, 0, stream>>>(rsum, 0.f, 1);
  k_redsum<<<grid1((size_t)Nn), 256, 0, stream>>>(logits, rmax, rsum, Nn);
  k_fill_f32<<<grid1(HID), 256, 0, stream>>>(pooled, 0.f, HID);
  k_pool<<<(Nn + 63) / 64, 256, 0, stream>>>(x, logits, rmax, rsum, pooled, Nn);
  k_mlp1<<<grid1(2 * LAT), 256, 0, stream>>>(pooled, p1_w, p1_b, h1);
  k_mlp2<<<grid1(LAT), 256, 0, stream>>>(h1, p2_w, p2_b, zbuf);
  k_ln<0><<<1, 256, 0, stream>>>(zbuf, pn_g, pn_b, (float*)d_out, LAT);
}
